// Block_32762010534491
// MI455X (gfx1250) — matrix operand order repaired and checked by reading, not compiled, not run on hardware
//
#include <hip/hip_runtime.h>
#include <hip/hip_bf16.h>

// Transformer block fwd for MI455X (gfx1250, wave32, WMMA).
// All GEMMs via v_wmma_f32_16x16x32_bf16 (bf16 in, fp32 accumulate).
// B operands stored pre-transposed so every WMMA fragment is two contiguous
// per-lane 16B runs (2x global_load_b128), no strided gathers.
// Softmax / LayerNorm / residuals stay fp32.

#define B_  4
#define T_  2048
#define D_  256
#define H_  8
#define DK_ 32
#define F_  1024   // 4*D

typedef __attribute__((ext_vector_type(16))) __bf16 v16bf;
typedef __attribute__((ext_vector_type(8)))  float  v8f;

__device__ __forceinline__ unsigned short f32_to_bf16_rn(float f) {
  unsigned int u = __float_as_uint(f);
  unsigned int lsb = (u >> 16) & 1u;
  u += 0x7fffu + lsb;                 // round-to-nearest-even
  return (unsigned short)(u >> 16);
}

union FragBF {
  unsigned short u[16];
  uint4          q[2];
  v16bf          v;
};

// ---- A fragment (16x32 bf16, row-major A, lda in elements) ----
// lane L, elem e: M = m0 + (L&15); K = k0 + (e%8) + 8*(L>>4) + 16*(e/8)
// -> per lane: two contiguous 8-element runs => two 16B loads.
__device__ __forceinline__ v16bf load_fragA_bf16(const unsigned short* A, int lda,
                                                 int m0, int k0, int lane) {
  FragBF fa;
  const uint4* p = (const uint4*)(A + (size_t)(m0 + (lane & 15)) * lda
                                    + k0 + 8 * (lane >> 4));
  fa.q[0] = p[0];
  fa.q[1] = p[2];   // +16 bf16 elements
  return fa.v;
}

// ---- B fragment from TRANSPOSED B (Bt[n][k], ldbt in elements) ----
// lane L, elem e: K = k0 + 16*(L>>4) + e; N = n0 + (L&15)
// -> per lane: one contiguous 16-element run => two 16B loads.
__device__ __forceinline__ v16bf load_fragBT_bf16(const unsigned short* Bt, int ldbt,
                                                  int k0, int n0, int lane) {
  FragBF fb;
  const uint4* p = (const uint4*)(Bt + (size_t)(n0 + (lane & 15)) * ldbt
                                     + k0 + 16 * (lane >> 4));
  fb.q[0] = p[0];
  fb.q[1] = p[1];
  return fb.v;
}

// ------------- fp32 -> bf16 weight convert + transpose -------------
// src: [K,N] row-major fp32 ; dT: [N,K] row-major bf16
__global__ void f2bT_kernel(const float* __restrict__ s, unsigned short* __restrict__ dT,
                            int K, int N) {
  int i = blockIdx.x * blockDim.x + threadIdx.x;
  if (i >= K * N) return;
  int n = i % N, k = i / N;
  dT[(size_t)n * K + k] = f32_to_bf16_rn(s[i]);
}

// --------- V[b,t,h*32+d] (bf16) -> Vt[b,h,d,t] (bf16) ---------
__global__ void vtrans_kernel(const unsigned short* __restrict__ Vb,
                              unsigned short* __restrict__ Vt) {
  int idx = blockIdx.x * blockDim.x + threadIdx.x;   // over B*T*D = 2^21
  int d = idx & (D_ - 1);
  int t = (idx >> 8) & (T_ - 1);
  int b = idx >> 19;
  int h = d >> 5, dk = d & 31;
  Vt[(((size_t)b * H_ + h) * DK_ + dk) * T_ + t] = Vb[idx];
}

// ---------------- LayerNorm (wave per row of 256) ----------------
__global__ void ln_kernel(const float* __restrict__ x, const float* __restrict__ g,
                          const float* __restrict__ b,
                          float* __restrict__ outF, unsigned short* __restrict__ outB,
                          int rows) {
  int lane = threadIdx.x & 31;
  int row  = blockIdx.x * (blockDim.x >> 5) + (threadIdx.x >> 5);
  if (row >= rows) return;
  const float* xr = x + (size_t)row * D_;
  float v[8]; float s = 0.f;
#pragma unroll
  for (int i = 0; i < 8; ++i) { v[i] = xr[lane + 32 * i]; s += v[i]; }
#pragma unroll
  for (int o = 16; o > 0; o >>= 1) s += __shfl_xor(s, o, 32);
  float mu = s * (1.f / D_);
  float s2 = 0.f;
#pragma unroll
  for (int i = 0; i < 8; ++i) { float d = v[i] - mu; s2 += d * d; }
#pragma unroll
  for (int o = 16; o > 0; o >>= 1) s2 += __shfl_xor(s2, o, 32);
  float inv = rsqrtf(s2 * (1.f / D_) + 1e-5f);
#pragma unroll
  for (int i = 0; i < 8; ++i) {
    int c = lane + 32 * i;
    float y = (v[i] - mu) * inv * g[c] + b[c];
    if (outF) outF[(size_t)row * D_ + c] = y;
    if (outB) outB[(size_t)row * D_ + c] = f32_to_bf16_rn(y);
  }
}

// -------- fused: y=LN(pre,g=1,b=0); x1=x+y; h=LN(x1,g2,b2) --------
__global__ void lnadd_kernel(const float* __restrict__ pre, const float* __restrict__ x,
                             const float* __restrict__ g2, const float* __restrict__ b2,
                             float* __restrict__ x1F, unsigned short* __restrict__ hB,
                             int rows) {
  int lane = threadIdx.x & 31;
  int row  = blockIdx.x * (blockDim.x >> 5) + (threadIdx.x >> 5);
  if (row >= rows) return;
  const float* pr = pre + (size_t)row * D_;
  const float* xr = x   + (size_t)row * D_;
  float v[8]; float s = 0.f;
#pragma unroll
  for (int i = 0; i < 8; ++i) { v[i] = pr[lane + 32 * i]; s += v[i]; }
#pragma unroll
  for (int o = 16; o > 0; o >>= 1) s += __shfl_xor(s, o, 32);
  float mu = s * (1.f / D_);
  float s2 = 0.f;
#pragma unroll
  for (int i = 0; i < 8; ++i) { float d = v[i] - mu; s2 += d * d; }
#pragma unroll
  for (int o = 16; o > 0; o >>= 1) s2 += __shfl_xor(s2, o, 32);
  float inv = rsqrtf(s2 * (1.f / D_) + 1e-5f);
  float x1[8]; float s1 = 0.f;
#pragma unroll
  for (int i = 0; i < 8; ++i) {
    float y = (v[i] - mu) * inv;               // gamma=1, beta=0
    x1[i] = xr[lane + 32 * i] + y;
    s1 += x1[i];
  }
#pragma unroll
  for (int o = 16; o > 0; o >>= 1) s1 += __shfl_xor(s1, o, 32);
  float mu1 = s1 * (1.f / D_);
  float s12 = 0.f;
#pragma unroll
  for (int i = 0; i < 8; ++i) { float d = x1[i] - mu1; s12 += d * d; }
#pragma unroll
  for (int o = 16; o > 0; o >>= 1) s12 += __shfl_xor(s12, o, 32);
  float inv1 = rsqrtf(s12 * (1.f / D_) + 1e-5f);
#pragma unroll
  for (int i = 0; i < 8; ++i) {
    int c = lane + 32 * i;
    x1F[(size_t)row * D_ + c] = x1[i];
    float hh = (x1[i] - mu1) * inv1 * g2[c] + b2[c];
    hB[(size_t)row * D_ + c] = f32_to_bf16_rn(hh);
  }
}

// ---------------- generic 16x16-tile WMMA GEMM ----------------
// C[M,N] = act(A[M,K]_bf16 @ Bt[N,K]^T_bf16 + bias) + residual
__global__ void gemm_kernel(const unsigned short* __restrict__ A, int lda,
                            const unsigned short* __restrict__ Bt, int ldbt,
                            const float* __restrict__ bias, const float* __restrict__ residual,
                            float* __restrict__ outF, unsigned short* __restrict__ outB,
                            int M, int N, int K, int ldc, int act) {
  int lane = threadIdx.x & 31;
  int wave = blockIdx.x * (blockDim.x >> 5) + (threadIdx.x >> 5);
  int tilesN = N >> 4;
  int totalTiles = (M >> 4) * tilesN;
  if (wave >= totalTiles) return;           // wave-uniform: EXEC stays all-ones
  int m0 = (wave / tilesN) << 4;
  int n0 = (wave % tilesN) << 4;
  const unsigned short* pA = A  + (size_t)(m0 + (lane & 15)) * lda  + 8  * (lane >> 4);
  const unsigned short* pB = Bt + (size_t)(n0 + (lane & 15)) * ldbt + 8  * (lane >> 4);
  v8f acc = {};
  for (int k0 = 0; k0 < K; k0 += 32, pA += 32, pB += 32) {
    __builtin_prefetch(pA + 32, 0, 0);      // -> global_prefetch_b8
    __builtin_prefetch(pB + 32, 0, 0);
    FragBF fa, fb;
    const uint4* a4 = (const uint4*)pA;
    fa.q[0] = a4[0]; fa.q[1] = a4[2];
    const uint4* b4 = (const uint4*)pB;
    fb.q[0] = b4[0]; fb.q[1] = b4[2];
    acc = __builtin_amdgcn_wmma_f32_16x16x32_bf16(false, fa.v, false, fb.v,
                                                  (short)0, acc, false, false);
  }
  int n = n0 + (lane & 15);
  float bn = bias ? bias[n] : 0.f;
#pragma unroll
  for (int r = 0; r < 8; ++r) {
    int m = m0 + r + 8 * (lane >> 4);
    float v = acc[r] + bn;
    if (act) v = 0.5f * v * (1.f + erff(v * 0.70710678118654752f));  // exact GELU
    if (residual) v += residual[(size_t)m * ldc + n];
    if (outF) outF[(size_t)m * ldc + n] = v;
    if (outB) outB[(size_t)m * ldc + n] = f32_to_bf16_rn(v);
  }
}

// -------- scores = mask ? -1e9 : (Q K^T)/sqrt(DK), one WMMA/tile --------
__global__ void scores_kernel(const unsigned short* __restrict__ Qb,
                              const unsigned short* __restrict__ Kb,
                              const unsigned char* __restrict__ mask,   // jnp bool = 1 byte
                              float* __restrict__ attn) {
  int lane = threadIdx.x & 31;
  int wave = blockIdx.x * (blockDim.x >> 5) + (threadIdx.x >> 5);
  int kt = wave & 127;
  int qt = (wave >> 7) & 127;
  int h  = (wave >> 14) & 7;
  int b  = wave >> 17;
  int q0 = qt << 4, kb0 = kt << 4;
  const unsigned short* Qbase = Qb + ((size_t)b * T_) * D_ + h * DK_;
  v16bf a = load_fragA_bf16(Qbase, D_, q0, 0, lane);     // 16x32: entire DK in one frag
  // B = K^T tile: B[r][n] = K[kb0+n][d=r] -> contiguous along head dim per lane
  FragBF fb;
  {
    int n = lane & 15;
    const uint4* p = (const uint4*)(Kb + ((size_t)(b * T_ + kb0 + n)) * D_
                                       + h * DK_ + 8 * (lane >> 4));
    fb.q[0] = p[0]; fb.q[1] = p[2];
  }
  v8f acc = {};
  acc = __builtin_amdgcn_wmma_f32_16x16x32_bf16(false, a, false, fb.v,
                                                (short)0, acc, false, false);
  const float sc = 0.17677669529663688f;   // 1/sqrt(32)
  int kcol = kb0 + (lane & 15);
  float* obase = attn + ((size_t)(b * H_ + h) * T_) * T_;
#pragma unroll
  for (int r = 0; r < 8; ++r) {
    int q = q0 + r + 8 * (lane >> 4);
    float v = acc[r] * sc;
    if (mask[((size_t)b * T_ + q) * T_ + kcol]) v = -1e9f;
    obase[(size_t)q * T_ + kcol] = v;
  }
}

// ---------------- row softmax over T=2048, in place ----------------
__global__ void softmax_kernel(float* __restrict__ attn) {
  __shared__ float sh[8];
  float* p = attn + (size_t)blockIdx.x * T_;
  int t = threadIdx.x;
  int w = t >> 5;
  float v[8]; float m = -3.402823466e38f;
#pragma unroll
  for (int i = 0; i < 8; ++i) { v[i] = p[t + 256 * i]; m = fmaxf(m, v[i]); }
#pragma unroll
  for (int o = 16; o > 0; o >>= 1) m = fmaxf(m, __shfl_xor(m, o, 32));
  if ((t & 31) == 0) sh[w] = m;
  __syncthreads();
  m = fmaxf(fmaxf(fmaxf(sh[0], sh[1]), fmaxf(sh[2], sh[3])),
            fmaxf(fmaxf(sh[4], sh[5]), fmaxf(sh[6], sh[7])));
  __syncthreads();
  float s = 0.f;
#pragma unroll
  for (int i = 0; i < 8; ++i) { v[i] = __expf(v[i] - m); s += v[i]; }
#pragma unroll
  for (int o = 16; o > 0; o >>= 1) s += __shfl_xor(s, o, 32);
  if ((t & 31) == 0) sh[w] = s;
  __syncthreads();
  s = sh[0] + sh[1] + sh[2] + sh[3] + sh[4] + sh[5] + sh[6] + sh[7];
  float inv = 1.f / s;
#pragma unroll
  for (int i = 0; i < 8; ++i) p[t + 256 * i] = v[i] * inv;
}

// -------- ctx[16,32] = attn[16,T] @ V[T,32], 2 WMMA per 32-K step --------
// V supplied transposed: Vt[b,h,d,t] -> contiguous B fragments.
__global__ void ctx_kernel(const float* __restrict__ attn,
                           const unsigned short* __restrict__ Vt,
                           unsigned short* __restrict__ ctxb) {
  int lane = threadIdx.x & 31;
  int wave = blockIdx.x * (blockDim.x >> 5) + (threadIdx.x >> 5);
  int qt = wave & 127;
  int h  = (wave >> 7) & 7;
  int b  = wave >> 10;
  int q0 = qt << 4;
  const float* arow = attn + ((size_t)(b * H_ + h) * T_ + q0 + (lane & 15)) * T_
                           + 8 * (lane >> 4);
  const unsigned short* pB0 = Vt + (((size_t)b * H_ + h) * DK_ + (lane & 15)) * T_
                                 + 8 * (lane >> 4);
  const unsigned short* pB1 = pB0 + 16 * (size_t)T_;   // n0 = 16 half of head dim
  v8f acc0 = {}, acc1 = {};
  for (int kb = 0; kb < T_; kb += 32, arow += 32, pB0 += 32, pB1 += 32) {
    __builtin_prefetch(arow + 32, 0, 0);
    __builtin_prefetch(pB0 + 32, 0, 0);
    __builtin_prefetch(pB1 + 32, 0, 0);
    FragBF fa, fb0, fb1;
    float4 f0 = *(const float4*)(arow);
    float4 f1 = *(const float4*)(arow + 4);
    float4 f2 = *(const float4*)(arow + 16);
    float4 f3 = *(const float4*)(arow + 20);
    float tf[16] = {f0.x, f0.y, f0.z, f0.w, f1.x, f1.y, f1.z, f1.w,
                    f2.x, f2.y, f2.z, f2.w, f3.x, f3.y, f3.z, f3.w};
#pragma unroll
    for (int e = 0; e < 16; ++e) fa.u[e] = f32_to_bf16_rn(tf[e]);
    const uint4* b40 = (const uint4*)pB0;
    const uint4* b41 = (const uint4*)pB1;
    fb0.q[0] = b40[0]; fb0.q[1] = b40[2];
    fb1.q[0] = b41[0]; fb1.q[1] = b41[2];
    acc0 = __builtin_amdgcn_wmma_f32_16x16x32_bf16(false, fa.v, false, fb0.v,
                                                   (short)0, acc0, false, false);
    acc1 = __builtin_amdgcn_wmma_f32_16x16x32_bf16(false, fa.v, false, fb1.v,
                                                   (short)0, acc1, false, false);
  }
  int d0 = lane & 15;
#pragma unroll
  for (int r = 0; r < 8; ++r) {
    int q = q0 + r + 8 * (lane >> 4);
    size_t base = ((size_t)b * T_ + q) * D_ + h * DK_;
    ctxb[base + d0]      = f32_to_bf16_rn(acc0[r]);
    ctxb[base + 16 + d0] = f32_to_bf16_rn(acc1[r]);
  }
}

extern "C" void kernel_launch(void* const* d_in, const int* in_sizes, int n_in,
                              void* d_out, int out_size, void* d_ws, size_t ws_size,
                              hipStream_t stream) {
  (void)in_sizes; (void)n_in; (void)out_size; (void)ws_size;
  const float* x          = (const float*)d_in[0];
  const unsigned char* mk = (const unsigned char*)d_in[1];   // bool tensor, 1 byte/elem
  const float* ln1_w = (const float*)d_in[2];
  const float* ln1_b = (const float*)d_in[3];
  const float* ln2_w = (const float*)d_in[4];
  const float* ln2_b = (const float*)d_in[5];
  const float* Wq = (const float*)d_in[6];
  const float* Wk = (const float*)d_in[7];
  const float* Wv = (const float*)d_in[8];
  const float* Wo = (const float*)d_in[9];
  const float* W1 = (const float*)d_in[10];
  const float* b1 = (const float*)d_in[11];
  const float* W2 = (const float*)d_in[12];
  const float* b2 = (const float*)d_in[13];

  const int rows = B_ * T_;   // 8192

  // ---- workspace carve-out (~70 MB) ----
  char* ws = (char*)d_ws;
  size_t off = 0;
  auto carve = [&](size_t bytes) -> void* {
    void* p = ws + off; off += (bytes + 255) & ~(size_t)255; return p;
  };
  unsigned short* WqT = (unsigned short*)carve((size_t)D_ * D_ * 2);   // [N,K]
  unsigned short* WkT = (unsigned short*)carve((size_t)D_ * D_ * 2);
  unsigned short* WvT = (unsigned short*)carve((size_t)D_ * D_ * 2);
  unsigned short* WoT = (unsigned short*)carve((size_t)D_ * D_ * 2);
  unsigned short* W1T = (unsigned short*)carve((size_t)D_ * F_ * 2);   // [F,D]
  unsigned short* W2T = (unsigned short*)carve((size_t)F_ * D_ * 2);   // [D,F]
  float*          xxF = (float*)         carve((size_t)rows * D_ * 4);
  unsigned short* xxB = (unsigned short*)carve((size_t)rows * D_ * 2);
  unsigned short* Qb  = (unsigned short*)carve((size_t)rows * D_ * 2);
  unsigned short* Kb  = (unsigned short*)carve((size_t)rows * D_ * 2);
  unsigned short* Vb  = (unsigned short*)carve((size_t)rows * D_ * 2);
  unsigned short* Vt  = (unsigned short*)carve((size_t)rows * D_ * 2); // [B,H,DK,T]
  unsigned short* ctxb= (unsigned short*)carve((size_t)rows * D_ * 2);
  float*          preF= (float*)         carve((size_t)rows * D_ * 4);
  float*          x1F = (float*)         carve((size_t)rows * D_ * 4);
  unsigned short* hB  = (unsigned short*)carve((size_t)rows * D_ * 2);
  unsigned short* gB  = (unsigned short*)carve((size_t)rows * F_ * 2);

  float* out0 = (float*)d_out;                       // [B,T,D]
  float* attn = out0 + (size_t)rows * D_;            // [B,H,T,T]

  // 1) weights -> bf16, transposed
  f2bT_kernel<<<(D_ * D_ + 255) / 256, 256, 0, stream>>>(Wq, WqT, D_, D_);
  f2bT_kernel<<<(D_ * D_ + 255) / 256, 256, 0, stream>>>(Wk, WkT, D_, D_);
  f2bT_kernel<<<(D_ * D_ + 255) / 256, 256, 0, stream>>>(Wv, WvT, D_, D_);
  f2bT_kernel<<<(D_ * D_ + 255) / 256, 256, 0, stream>>>(Wo, WoT, D_, D_);
  f2bT_kernel<<<(D_ * F_ + 255) / 256, 256, 0, stream>>>(W1, W1T, D_, F_);
  f2bT_kernel<<<(F_ * D_ + 255) / 256, 256, 0, stream>>>(W2, W2T, F_, D_);

  // 2) xx = LN1(x)
  ln_kernel<<<rows / 8, 256, 0, stream>>>(x, ln1_w, ln1_b, xxF, xxB, rows);

  // 3) Q/K/V projections
  {
    int tiles = (rows / 16) * (D_ / 16);
    gemm_kernel<<<tiles / 4, 128, 0, stream>>>(xxB, D_, WqT, D_, nullptr, nullptr,
                                               nullptr, Qb, rows, D_, D_, D_, 0);
    gemm_kernel<<<tiles / 4, 128, 0, stream>>>(xxB, D_, WkT, D_, nullptr, nullptr,
                                               nullptr, Kb, rows, D_, D_, D_, 0);
    gemm_kernel<<<tiles / 4, 128, 0, stream>>>(xxB, D_, WvT, D_, nullptr, nullptr,
                                               nullptr, Vb, rows, D_, D_, D_, 0);
  }

  // 3b) V -> Vt[b,h,d,t]
  vtrans_kernel<<<(rows * D_) / 256, 256, 0, stream>>>(Vb, Vt);

  // 4) masked scaled scores -> d_out attn region
  {
    int waves = B_ * H_ * (T_ / 16) * (T_ / 16);     // 524288
    scores_kernel<<<waves / 4, 128, 0, stream>>>(Qb, Kb, mk, attn);
  }

  // 5) softmax rows
  softmax_kernel<<<B_ * H_ * T_, 256, 0, stream>>>(attn);

  // 6) ctx = attn @ V
  ctx_kernel<<<(B_ * H_ * (T_ / 16)) / 4, 128, 0, stream>>>(attn, Vt, ctxb);

  // 7) pre = ctx @ Wo + xx
  gemm_kernel<<<((rows / 16) * (D_ / 16)) / 4, 128, 0, stream>>>(
      ctxb, D_, WoT, D_, nullptr, xxF, preF, nullptr, rows, D_, D_, D_, 0);

  // 8) y = LN(pre); x1 = x + y; h = LN2(x1)
  lnadd_kernel<<<rows / 8, 256, 0, stream>>>(preF, x, ln2_w, ln2_b, x1F, hB, rows);

  // 9) g = gelu(h @ W1 + b1)
  gemm_kernel<<<((rows / 16) * (F_ / 16)) / 4, 128, 0, stream>>>(
      hB, D_, W1T, D_, b1, nullptr, nullptr, gB, rows, F_, D_, F_, 1);

  // 10) out0 = g @ W2 + b2 + x1
  gemm_kernel<<<((rows / 16) * (D_ / 16)) / 4, 128, 0, stream>>>(
      gB, F_, W2T, F_, b2, x1F, out0, nullptr, rows, D_, F_, D_, 0);
}
